// HeteroActor_31267361915424
// MI455X (gfx1250) — matrix operand order repaired and checked by reading, not compiled, not run on hardware
//
#include <hip/hip_runtime.h>
#include <hip/hip_bf16.h>
#include <math.h>

#define N_P   50000
#define N_A   50000
#define NE    1600000
#define INF_ENC 0x007FFFFFu   // ordered-uint encoding of -inf

typedef __attribute__((ext_vector_type(16))) _Float16 v16h;
typedef __attribute__((ext_vector_type(8)))  float    v8f;

// ---- monotonic float <-> uint encoding for atomicMax on floats ----
__device__ __forceinline__ unsigned fenc(float f) {
  unsigned u = __float_as_uint(f);
  return (u & 0x80000000u) ? ~u : (u | 0x80000000u);
}
__device__ __forceinline__ float fdec(unsigned e) {
  unsigned u = (e & 0x80000000u) ? (e ^ 0x80000000u) : ~e;
  return __uint_as_float(u);
}

__global__ void fill_u32_kernel(unsigned* __restrict__ p, unsigned v, int n) {
  int i = blockIdx.x * blockDim.x + threadIdx.x;
  if (i < n) p[i] = v;
}

// =====================================================================
// WMMA GEMM: C[M,64] = (w0*A0 + w1*A1)[M,K] @ W[K,64] + bias, optional relu.
// f32 in -> f16 fragments -> v_wmma_f32_16x16x32_f16 -> f32 out.
// Block = 128 threads = 4 waves; wave handles 16 rows x all 64 cols.
// =====================================================================
template <int K>
__global__ void wmma_gemm_kernel(const float* __restrict__ A0,
                                 const float* __restrict__ A1,
                                 const float* __restrict__ attn,   // null -> identity
                                 const float* __restrict__ W,      // [K][64] row-major
                                 const float* __restrict__ bias,   // [64]
                                 float* __restrict__ C, int M, int doRelu) {
  constexpr int LK = K + 8;                 // padded LDS row (halves)
  extern __shared__ _Float16 smem[];
  _Float16* WT = smem;                      // [64][LK] : WT[n][k] (transposed)
  const int tid  = threadIdx.x;
  const int lane = tid & 31;
  const int wave = tid >> 5;
  _Float16* AL = smem + 64 * LK + wave * 16 * LK;   // per-wave [16][LK]

  const float w0 = attn ? attn[0] : 1.0f;
  const float w1 = (attn && A1) ? attn[1] : 0.0f;

  // Stage W transposed into LDS (f32 -> f16).
  for (int e = tid; e < K * 64; e += blockDim.x) {
    int k = e >> 6, n = e & 63;
    WT[n * LK + k] = (_Float16)W[e];
  }

  // Stage this wave's 16 x K A tile (fused attn-combine), f32 -> f16.
  const int row0 = blockIdx.x * 64 + wave * 16;
  for (int idx = lane; idx < 16 * (K / 4); idx += 32) {
    int r  = idx / (K / 4);
    int c4 = (idx % (K / 4)) * 4;
    int row  = row0 + r;
    int rowc = row < M ? row : (M - 1);
    float4 a = *(const float4*)(A0 + (size_t)rowc * K + c4);
    float vx = w0 * a.x, vy = w0 * a.y, vz = w0 * a.z, vw = w0 * a.w;
    if (A1) {
      float4 b = *(const float4*)(A1 + (size_t)rowc * K + c4);
      vx += w1 * b.x; vy += w1 * b.y; vz += w1 * b.z; vw += w1 * b.w;
    }
    _Float16* dst = AL + r * LK + c4;
    dst[0] = (_Float16)vx; dst[1] = (_Float16)vy;
    dst[2] = (_Float16)vz; dst[3] = (_Float16)vw;
  }
  __syncthreads();

  const int m    = lane & 15;   // column-within-tile (B/C/D) / row (A)
  const int hsel = lane >> 4;   // lane half-select
  v8f acc[4] = {};              // 4 N-tiles of 16 -> full 64 cols

  const _Float16* ALm = AL + m * LK;
  #pragma unroll
  for (int kk = 0; kk < K; kk += 32) {
    // A fragment: ISA 16-bit A 16x32 layout.
    v16h a;
    const int aK0 = kk + hsel * 8;
    #pragma unroll
    for (int i = 0; i < 8; ++i) {
      a[i]     = ALm[aK0 + i];
      a[8 + i] = ALm[aK0 + 16 + i];
    }
    const int bK0 = kk + hsel * 8;
    #pragma unroll
    for (int t = 0; t < 4; ++t) {
      const _Float16* wr = WT + (t * 16 + m) * LK + bK0;
      v16h b;
      #pragma unroll
      for (int i = 0; i < 8; ++i) { b[i] = wr[i]; b[8 + i] = wr[16 + i]; }
      acc[t] = __builtin_amdgcn_wmma_f32_16x16x32_f16(
          false, a, false, b, (short)0, acc[t], false, false);
    }
  }

  // Epilogue: C/D layout VGPR r -> row r + hsel*8, lane%16 -> col.
  #pragma unroll
  for (int t = 0; t < 4; ++t) {
    const int n  = t * 16 + m;
    const float bv = bias[n];
    #pragma unroll
    for (int r = 0; r < 8; ++r) {
      int row = row0 + hsel * 8 + r;
      if (row < M) {
        float v = acc[t][r] + bv;
        if (doRelu) v = fmaxf(v, 0.0f);
        C[(size_t)row * 64 + n] = v;
      }
    }
  }
}

// ---- per-node attention scores s[n][h] = sum_d h[n][h][d] * a[h][d] ----
__global__ void node_score_kernel(const float* __restrict__ h,
                                  const float* __restrict__ avec,
                                  float* __restrict__ s, int N) {
  int n = blockIdx.x * blockDim.x + threadIdx.x;
  if (n >= N) return;
  const float* hr = h + (size_t)n * 64;
  #pragma unroll
  for (int hh = 0; hh < 4; ++hh) {
    float acc = 0.f;
    #pragma unroll
    for (int d = 0; d < 16; ++d) acc += hr[hh * 16 + d] * avec[hh * 16 + d];
    s[n * 4 + hh] = acc;
  }
}

__device__ __forceinline__ float edge_alpha(const float* ss, const float* sd,
                                            int si, int di, int hh) {
  float a = ss[si * 4 + hh] + sd[di * 4 + hh];
  return a < 0.f ? a * 0.2f : a;   // leaky_relu, slope 0.2
}

// ---- pass 1: segment max of alpha per (dst, head) ----
__global__ void edge_max_kernel(const int* __restrict__ src, const int* __restrict__ dst,
                                const float* __restrict__ ss, const float* __restrict__ sd,
                                unsigned* __restrict__ amax, int E4) {
  int i = blockIdx.x * blockDim.x + threadIdx.x;
  if (i >= E4) return;
  int e = i >> 2, hh = i & 3;
  int si = src[e], di = dst[e];
  float a = edge_alpha(ss, sd, si, di, hh);
  atomicMax(&amax[di * 4 + hh], fenc(a));
}

// ---- pass 2: segment sum of exp(alpha - amax) ----
__global__ void edge_den_kernel(const int* __restrict__ src, const int* __restrict__ dst,
                                const float* __restrict__ ss, const float* __restrict__ sd,
                                const unsigned* __restrict__ amax,
                                float* __restrict__ den, int E4) {
  int i = blockIdx.x * blockDim.x + threadIdx.x;
  if (i >= E4) return;
  int e = i >> 2, hh = i & 3;
  int si = src[e], di = dst[e];
  float a  = edge_alpha(ss, sd, si, di, hh);
  float ex = __expf(a - fdec(amax[di * 4 + hh]));
  atomicAdd(&den[di * 4 + hh], ex);
}

// ---- pass 3: weighted message scatter (16 floats per (edge,head) thread) ----
__global__ void edge_scatter_kernel(const int* __restrict__ src, const int* __restrict__ dst,
                                    const float* __restrict__ ss, const float* __restrict__ sd,
                                    const unsigned* __restrict__ amax,
                                    const float* __restrict__ den,
                                    const float* __restrict__ hsrc,
                                    float* __restrict__ outp, int E4) {
  int i = blockIdx.x * blockDim.x + threadIdx.x;
  if (i >= E4) return;
  int e = i >> 2, hh = i & 3;
  int si = src[e], di = dst[e];
  float a  = edge_alpha(ss, sd, si, di, hh);
  float ex = __expf(a - fdec(amax[di * 4 + hh]));
  float w  = ex / (den[di * 4 + hh] + 1e-16f);
  const float4* hs = (const float4*)(hsrc + (size_t)si * 64 + hh * 16);
  float* op = outp + (size_t)di * 64 + hh * 16;
  #pragma unroll
  for (int q4 = 0; q4 < 4; ++q4) {
    float4 v = hs[q4];
    atomicAdd(&op[q4 * 4 + 0], v.x * w);
    atomicAdd(&op[q4 * 4 + 1], v.y * w);
    atomicAdd(&op[q4 * 4 + 2], v.z * w);
    atomicAdd(&op[q4 * 4 + 3], v.w * w);
  }
}

__global__ void relu_kernel(float* __restrict__ p, int n) {
  int i = blockIdx.x * blockDim.x + threadIdx.x;
  if (i < n) p[i] = fmaxf(p[i], 0.0f);
}

// ---- semantic attention: accum[j] += sum_n tanh((out @ Wk + bk)[n, j]) ----
__global__ void semantic_partial_kernel(const float* __restrict__ outp,
                                        const float* __restrict__ Wk,
                                        const float* __restrict__ bk,
                                        float* __restrict__ accum, int N) {
  __shared__ float sWk[64 * 64];
  for (int e = threadIdx.x; e < 64 * 64; e += blockDim.x) sWk[e] = Wk[e];
  __syncthreads();
  int n = blockIdx.x * blockDim.x + threadIdx.x;
  int rc = n < N ? n : 0;
  float valid = n < N ? 1.f : 0.f;
  float row[64];
  #pragma unroll
  for (int k = 0; k < 64; ++k) row[k] = outp[(size_t)rc * 64 + k];
  for (int j = 0; j < 64; ++j) {
    float acc = bk[j];
    #pragma unroll
    for (int k = 0; k < 64; ++k) acc += row[k] * sWk[k * 64 + j];
    float t = tanhf(acc) * valid;
    #pragma unroll
    for (int off = 16; off > 0; off >>= 1) t += __shfl_xor(t, off, 32);
    if ((threadIdx.x & 31) == 0) atomicAdd(&accum[j], t);
  }
}

// ---- softmax over the 2 edge-type scores ----
__global__ void semantic_attn_kernel(const float* __restrict__ accum,
                                     const float* __restrict__ q,
                                     float* __restrict__ attn, float invN) {
  __shared__ float s0, s1;
  if (threadIdx.x == 0) { s0 = 0.f; s1 = 0.f; }
  __syncthreads();
  int j = threadIdx.x;   // 64 threads
  atomicAdd(&s0, q[j] * accum[j]      * invN);
  atomicAdd(&s1, q[j] * accum[64 + j] * invN);
  __syncthreads();
  if (threadIdx.x == 0) {
    float m = fmaxf(s0, s1);
    float e0 = __expf(s0 - m), e1 = __expf(s1 - m);
    float inv = 1.0f / (e0 + e1);
    attn[0] = e0 * inv;
    attn[1] = e1 * inv;
  }
}

// ---- decoder layer 2 (GEMV) + global max of z ----
__global__ void zdot_kernel(const float* __restrict__ hid, const float* __restrict__ Wd2,
                            const float* __restrict__ bd2, float* __restrict__ z,
                            unsigned* __restrict__ zmax, int N) {
  int n = blockIdx.x * blockDim.x + threadIdx.x;
  if (n >= N) return;
  const float* hr = hid + (size_t)n * 64;
  float acc = bd2[0];
  #pragma unroll
  for (int k = 0; k < 64; ++k) acc += hr[k] * Wd2[k];
  z[n] = acc;
  atomicMax(zmax, fenc(acc));
}

__global__ void zexp_kernel(const float* __restrict__ z, const unsigned* __restrict__ zmax,
                            float* __restrict__ zsum, float* __restrict__ out, int N) {
  int n = blockIdx.x * blockDim.x + threadIdx.x;
  if (n >= N) return;
  float ex = __expf(z[n] - fdec(*zmax));
  out[n] = ex;
  atomicAdd(zsum, ex);
}

__global__ void znorm_kernel(float* __restrict__ out, const float* __restrict__ zsum, int N) {
  int n = blockIdx.x * blockDim.x + threadIdx.x;
  if (n >= N) return;
  out[n] = out[n] / (*zsum);
}

// =====================================================================
extern "C" void kernel_launch(void* const* d_in, const int* in_sizes, int n_in,
                              void* d_out, int out_size, void* d_ws, size_t ws_size,
                              hipStream_t stream) {
  (void)in_sizes; (void)n_in; (void)out_size; (void)ws_size;
  const float* x_place   = (const float*)d_in[0];
  const float* x_atrans  = (const float*)d_in[1];
  const int*   e_pa_src  = (const int*)d_in[2];
  const int*   e_pa_dst  = (const int*)d_in[3];
  const int*   e_aa_src  = (const int*)d_in[4];
  const int*   e_aa_dst  = (const int*)d_in[5];
  // d_in[6..7] (e_ap_*) and d_in[16..17] (asrc_ap/adst_ap) only feed the
  // 'place' semantic output, which the reference never uses: dead code, skipped.
  const float* Wp_place  = (const float*)d_in[8];
  const float* bp_place  = (const float*)d_in[9];
  const float* Wp_atrans = (const float*)d_in[10];
  const float* bp_atrans = (const float*)d_in[11];
  const float* asrc_pa   = (const float*)d_in[12];
  const float* adst_pa   = (const float*)d_in[13];
  const float* asrc_aa   = (const float*)d_in[14];
  const float* adst_aa   = (const float*)d_in[15];
  const float* Wk        = (const float*)d_in[18];
  const float* bk        = (const float*)d_in[19];
  const float* qv        = (const float*)d_in[20];
  const float* Wd1       = (const float*)d_in[21];
  const float* bd1       = (const float*)d_in[22];
  const float* Wd2       = (const float*)d_in[23];
  const float* bd2       = (const float*)d_in[24];
  float* out = (float*)d_out;

  // ---- workspace carve-up (256B aligned; regions grouped for bulk init) ----
  char* ws = (char*)d_ws;
  size_t off = 0;
  auto take = [&](size_t bytes) -> char* {
    char* p = ws + off;
    off = (off + bytes + 255) & ~(size_t)255;
    return p;
  };
  float*    h_p      = (float*)take((size_t)N_P * 64 * 4);
  float*    h_a      = (float*)take((size_t)N_A * 64 * 4);
  float*    s_pa_src = (float*)take((size_t)N_P * 4 * 4);
  float*    s_pa_dst = (float*)take((size_t)N_A * 4 * 4);
  float*    s_aa_src = (float*)take((size_t)N_A * 4 * 4);
  float*    s_aa_dst = (float*)take((size_t)N_A * 4 * 4);
  // pattern-fill region: amax_pa | amax_aa | zmax  (encoded -inf)
  unsigned* amax_pa  = (unsigned*)take((size_t)N_A * 4 * 4);
  unsigned* amax_aa  = (unsigned*)take((size_t)N_A * 4 * 4);
  unsigned* zmaxp    = (unsigned*)take(256);
  size_t fill_words  = ((char*)zmaxp + 256 - (char*)amax_pa) / 4;
  // zero region: den_pa | den_aa | out_pa | out_aa | sem_accum | zsum
  char*  zero_base   = ws + off;
  float* den_pa      = (float*)take((size_t)N_A * 4 * 4);
  float* den_aa      = (float*)take((size_t)N_A * 4 * 4);
  float* out_pa      = (float*)take((size_t)N_A * 64 * 4);
  float* out_aa      = (float*)take((size_t)N_A * 64 * 4);
  float* sem_accum   = (float*)take(2 * 64 * 4);
  float* zsump       = (float*)take(256);
  size_t zero_len    = (size_t)((ws + off) - zero_base);
  float* attnp       = (float*)take(256);
  float* hid         = (float*)take((size_t)N_A * 64 * 4);
  float* zbuf        = (float*)take((size_t)N_A * 4);

  // ---- init accumulators (every call: harness does not re-poison) ----
  {
    int nw = (int)fill_words;
    fill_u32_kernel<<<(nw + 255) / 256, 256, 0, stream>>>(amax_pa, INF_ENC, nw);
  }
  hipMemsetAsync(zero_base, 0, zero_len, stream);

  // ---- node projections via WMMA: h = X @ Wp + bp ----
  {
    dim3 g((N_P + 63) / 64), b(128);
    size_t sh = (size_t)128 * (128 + 8) * sizeof(_Float16);  // 34816 B
    wmma_gemm_kernel<128><<<g, b, sh, stream>>>(x_place,  nullptr, nullptr,
                                                Wp_place,  bp_place,  h_p, N_P, 0);
    wmma_gemm_kernel<128><<<g, b, sh, stream>>>(x_atrans, nullptr, nullptr,
                                                Wp_atrans, bp_atrans, h_a, N_A, 0);
  }

  // ---- per-node alpha scores (one pass over h instead of per-edge gathers) ----
  int nb = (N_A + 255) / 256;
  node_score_kernel<<<nb, 256, 0, stream>>>(h_p, asrc_pa, s_pa_src, N_P);
  node_score_kernel<<<nb, 256, 0, stream>>>(h_a, adst_pa, s_pa_dst, N_A);
  node_score_kernel<<<nb, 256, 0, stream>>>(h_a, asrc_aa, s_aa_src, N_A);
  node_score_kernel<<<nb, 256, 0, stream>>>(h_a, adst_aa, s_aa_dst, N_A);

  // ---- edge attention: 3 passes x 2 live edge types ----
  const int E4 = NE * 4;
  const int eb = (E4 + 255) / 256;
  edge_max_kernel<<<eb, 256, 0, stream>>>(e_pa_src, e_pa_dst, s_pa_src, s_pa_dst, amax_pa, E4);
  edge_max_kernel<<<eb, 256, 0, stream>>>(e_aa_src, e_aa_dst, s_aa_src, s_aa_dst, amax_aa, E4);
  edge_den_kernel<<<eb, 256, 0, stream>>>(e_pa_src, e_pa_dst, s_pa_src, s_pa_dst, amax_pa, den_pa, E4);
  edge_den_kernel<<<eb, 256, 0, stream>>>(e_aa_src, e_aa_dst, s_aa_src, s_aa_dst, amax_aa, den_aa, E4);
  edge_scatter_kernel<<<eb, 256, 0, stream>>>(e_pa_src, e_pa_dst, s_pa_src, s_pa_dst,
                                              amax_pa, den_pa, h_p, out_pa, E4);
  edge_scatter_kernel<<<eb, 256, 0, stream>>>(e_aa_src, e_aa_dst, s_aa_src, s_aa_dst,
                                              amax_aa, den_aa, h_a, out_aa, E4);

  // relu over out_pa and out_aa (contiguous in workspace)
  relu_kernel<<<(2 * N_A * 64 + 255) / 256, 256, 0, stream>>>(out_pa, 2 * N_A * 64);

  // ---- semantic-level attention ----
  semantic_partial_kernel<<<nb, 256, 0, stream>>>(out_pa, Wk, bk, sem_accum,      N_A);
  semantic_partial_kernel<<<nb, 256, 0, stream>>>(out_aa, Wk, bk, sem_accum + 64, N_A);
  semantic_attn_kernel<<<1, 64, 0, stream>>>(sem_accum, qv, attnp, 1.0f / (float)N_A);

  // ---- decoder layer 1 via WMMA with fused x_a = attn0*out_pa + attn1*out_aa ----
  {
    dim3 g((N_A + 63) / 64), b(128);
    size_t sh = (size_t)128 * (64 + 8) * sizeof(_Float16);   // 18432 B
    wmma_gemm_kernel<64><<<g, b, sh, stream>>>(out_pa, out_aa, attnp,
                                               Wd1, bd1, hid, N_A, 1);
  }

  // ---- decoder layer 2 + softmax over nodes ----
  zdot_kernel<<<nb, 256, 0, stream>>>(hid, Wd2, bd2, zbuf, zmaxp, N_A);
  zexp_kernel<<<nb, 256, 0, stream>>>(zbuf, zmaxp, zsump, out, N_A);
  znorm_kernel<<<nb, 256, 0, stream>>>(out, zsump, N_A);
}
